// QuantizedLinear_48086453846078
// MI455X (gfx1250) — matrix operand order repaired and checked by reading, not compiled, not run on hardware
//
#include <hip/hip_runtime.h>

// ---- MI455X / gfx1250: wave32, WMMA 16x16x32 f16, double-buffered LDS ----
typedef _Float16 v16h __attribute__((ext_vector_type(16)));
typedef _Float16 v8h  __attribute__((ext_vector_type(8)));
typedef float    v8f  __attribute__((ext_vector_type(8)));
typedef float    v4f  __attribute__((ext_vector_type(4)));
typedef int      v4i  __attribute__((ext_vector_type(4)));

#define M_TOKENS 8192
#define IN_F     4096
#define OUT_F    4096

#define BM 128
#define BN 128
#define BK 32
#define LDS_STRIDE 40   // 32 + 8 halves pad; row pitch = 80B (16B aligned)

__global__ __launch_bounds__(256)
void quantized_linear_wmma_kernel(const float* __restrict__ x,
                                  const int*   __restrict__ wq,
                                  const float* __restrict__ scale,
                                  const float* __restrict__ zp,
                                  const float* __restrict__ bias,
                                  float*       __restrict__ out)
{
    // Double-buffered staging: 2 x (128x40 halves) for A and B = 40 KB LDS.
    __shared__ __align__(16) _Float16 As[2][BM * LDS_STRIDE];
    __shared__ __align__(16) _Float16 Bs[2][BN * LDS_STRIDE];

    const int tid  = threadIdx.x;
    const int lane = tid & 31;          // wave32
    const int wave = tid >> 5;          // 0..7
    const int m0   = blockIdx.y * BM;
    const int n0   = blockIdx.x * BN;

    // ---- staging assignment: thread t stages half a row (16 elems) ----
    const int srow = tid >> 1;          // 0..127
    const int scol = (tid & 1) * 16;    // 0 or 16
    const int soff = srow * LDS_STRIDE + scol;

    const float* xrow = x  + (size_t)(m0 + srow) * IN_F + scol;
    const int*   wrow = wq + (size_t)(n0 + srow) * IN_F + scol;
    const float  s    = scale[n0 + srow];   // per-out-channel dequant params
    const float  z    = zp[n0 + srow];

    // ---- accumulators: wave computes 16 rows x 128 cols = 8 WMMA tiles ----
    v8f acc[8];
    #pragma unroll
    for (int t = 0; t < 8; ++t)
        #pragma unroll
        for (int i = 0; i < 8; ++i)
            acc[t][i] = 0.0f;

    // ---- fragment addressing (ISA 16-bit A / B layouts, wave32) ----
    const int frow = lane & 15;
    const int ak0  = (lane < 16) ? 0 : 8;    // A: K {0..7,16..23} | {8..15,24..31}
    const int bk0  = (lane < 16) ? 0 : 8;    // B: K {0..7,16..23} | {8..15,24..31}
    const int aoff = (wave * 16 + frow) * LDS_STRIDE + ak0;

    // Staging registers (double buffer lives in regs between load & LDS store)
    v4f xa0, xa1, xa2, xa3;
    v4i qa0, qa1, qa2, qa3;

    auto load_tile = [&](int kb) {
        xa0 = *(const v4f*)(xrow + kb + 0);
        xa1 = *(const v4f*)(xrow + kb + 4);
        xa2 = *(const v4f*)(xrow + kb + 8);
        xa3 = *(const v4f*)(xrow + kb + 12);
        qa0 = *(const v4i*)(wrow + kb + 0);
        qa1 = *(const v4i*)(wrow + kb + 4);
        qa2 = *(const v4i*)(wrow + kb + 8);
        qa3 = *(const v4i*)(wrow + kb + 12);
    };

    auto store_tile = [&](int buf) {
        v8h h0 = { (_Float16)xa0[0], (_Float16)xa0[1], (_Float16)xa0[2], (_Float16)xa0[3],
                   (_Float16)xa1[0], (_Float16)xa1[1], (_Float16)xa1[2], (_Float16)xa1[3] };
        v8h h1 = { (_Float16)xa2[0], (_Float16)xa2[1], (_Float16)xa2[2], (_Float16)xa2[3],
                   (_Float16)xa3[0], (_Float16)xa3[1], (_Float16)xa3[2], (_Float16)xa3[3] };
        *(v8h*)(&As[buf][soff + 0]) = h0;
        *(v8h*)(&As[buf][soff + 8]) = h1;
        #define DQ(q) ((_Float16)(((float)(q) - z) * s))
        v8h g0 = { DQ(qa0[0]), DQ(qa0[1]), DQ(qa0[2]), DQ(qa0[3]),
                   DQ(qa1[0]), DQ(qa1[1]), DQ(qa1[2]), DQ(qa1[3]) };
        v8h g1 = { DQ(qa2[0]), DQ(qa2[1]), DQ(qa2[2]), DQ(qa2[3]),
                   DQ(qa3[0]), DQ(qa3[1]), DQ(qa3[2]), DQ(qa3[3]) };
        #undef DQ
        *(v8h*)(&Bs[buf][soff + 0]) = g0;
        *(v8h*)(&Bs[buf][soff + 8]) = g1;
    };

    auto compute_tile = [&](int buf) {
        const _Float16* aptr = &As[buf][aoff];
        v8h alo = *(const v8h*)(aptr);
        v8h ahi = *(const v8h*)(aptr + 16);
        v16h afrag = __builtin_shufflevector(alo, ahi,
            0,1,2,3,4,5,6,7,8,9,10,11,12,13,14,15);
        #pragma unroll
        for (int t = 0; t < 8; ++t) {
            const _Float16* bptr = &Bs[buf][(t * 16 + frow) * LDS_STRIDE + bk0];
            v8h blo = *(const v8h*)(bptr);
            v8h bhi = *(const v8h*)(bptr + 16);
            v16h bfrag = __builtin_shufflevector(blo, bhi,
                0,1,2,3,4,5,6,7,8,9,10,11,12,13,14,15);
            acc[t] = __builtin_amdgcn_wmma_f32_16x16x32_f16(
                /*neg_a=*/false, afrag, /*neg_b=*/false, bfrag,
                /*c_mod=*/(short)0, acc[t], /*reuse_a=*/false, /*reuse_b=*/false);
        }
    };

    const int KSTEPS = IN_F / BK;   // 128

    // ---- prologue: fill buffer 0 ----
    load_tile(0);
    store_tile(0);
    __syncthreads();

    // ---- steady state: 1 barrier / iteration; loads for kt+1 fly over WMMAs ----
    for (int kt = 0; kt < KSTEPS; ++kt) {
        const bool more = (kt + 1) < KSTEPS;
        if (more) {
            load_tile((kt + 1) * BK);
            if (kt + 2 < KSTEPS) {
                __builtin_prefetch(xrow + (kt + 2) * BK, 0, 1);  // global_prefetch_b8
                __builtin_prefetch(wrow + (kt + 2) * BK, 0, 1);
            }
        }

        compute_tile(kt & 1);            // read buffer kt&1

        if (more)
            store_tile((kt + 1) & 1);    // write the other buffer (no RAW w/ readers)

        __syncthreads();
    }

    // ---- epilogue: + bias, store fp32 (C/D layout: lane = col, vgpr = row) ----
    const int outrow0 = m0 + wave * 16 + (lane >> 4) * 8;
    #pragma unroll
    for (int t = 0; t < 8; ++t) {
        const int col = n0 + t * 16 + (lane & 15);
        const float b = bias[col];
        #pragma unroll
        for (int r = 0; r < 8; ++r) {
            out[(size_t)(outrow0 + r) * OUT_F + col] = acc[t][r] + b;
        }
    }
}

extern "C" void kernel_launch(void* const* d_in, const int* in_sizes, int n_in,
                              void* d_out, int out_size, void* d_ws, size_t ws_size,
                              hipStream_t stream) {
    (void)in_sizes; (void)n_in; (void)d_ws; (void)ws_size; (void)out_size;
    const float* x     = (const float*)d_in[0];
    const int*   wq    = (const int*)  d_in[1];
    const float* scale = (const float*)d_in[2];
    const float* zp    = (const float*)d_in[3];
    const float* bias  = (const float*)d_in[4];
    float*       out   = (float*)d_out;

    dim3 grid(OUT_F / BN, M_TOKENS / BM);   // (32, 64): n fastest -> A-tile reuse,
                                            // 67 MB weights stay L2-resident
    quantized_linear_wmma_kernel<<<grid, 256, 0, stream>>>(x, wq, scale, zp, bias, out);
}
